// LSTMAtt_10058813407972
// MI455X (gfx1250) — matrix operand order repaired and checked by reading, not compiled, not run on hardware
//
#include <hip/hip_runtime.h>

// ---------------------------------------------------------------------------
// Problem constants (from reference): T=128, B=32, V=32000, NI=512, NH=1024
// ---------------------------------------------------------------------------
#define TT  128
#define BB  32
#define VV  32000
#define NI_ 512
#define NH_ 1024
#define H4  4096   // 4*NH

typedef __attribute__((ext_vector_type(16))) __bf16 v16bf;
typedef __attribute__((ext_vector_type(8)))  __bf16 v8bf;
typedef __attribute__((ext_vector_type(8)))  float  v8f;

// ---------------------------------------------------------------------------
// bf16 WMMA GEMM:  C[M,N] = A[M,K] * W[N,K]^T  (+ bias[n]) (+ addRow[m,n])
// A row-major lda, W row-major ldb (K contiguous in both) -> direct WMMA
// fragment loads. Block = 256 thr = 8 wave32, block tile 32(M) x 256(N),
// each wave a 16x64 strip. Requires M%32==0, N%256==0, K%32==0 (true for all
// uses here), lda/ldb multiples of 16.
// ---------------------------------------------------------------------------
__global__ __launch_bounds__(256) void gemm_bf16_wmma(
    const __bf16* __restrict__ A, int lda,
    const __bf16* __restrict__ W, int ldb,
    const float* __restrict__ bias,              // nullable, [N]
    const float* __restrict__ addRow, int ldAdd, // nullable, [M, ldAdd]
    float* __restrict__ C, int ldc,
    int M, int N, int K)
{
    const int lane = threadIdx.x & 31;
    const int wave = threadIdx.x >> 5;
    const int m0 = blockIdx.y * 32 + (wave >> 2) * 16;
    const int n0 = blockIdx.x * 256 + (wave & 3) * 64;
    if (m0 >= M || n0 >= N) return;
    const int l16 = lane & 15;
    const int hi  = lane >> 4;          // 0: lanes 0-15, 1: lanes 16-31

    v8f acc[4] = {};

    // A fragment base: lanes 0-15 hold K {0..15}, lanes 16-31 {16..31}
    const __bf16* aRow = A + (size_t)(m0 + l16) * lda + hi * 16;

    for (int k0 = 0; k0 < K; k0 += 32) {
        v8bf alo = *(const v8bf*)(aRow + k0);        // 16B aligned
        v8bf ahi = *(const v8bf*)(aRow + k0 + 8);
        v16bf af = __builtin_shufflevector(alo, ahi,
                    0,1,2,3,4,5,6,7,8,9,10,11,12,13,14,15);
        #pragma unroll
        for (int s = 0; s < 4; ++s) {
            // B fragment: lane holds column n = n0+s*16+l16, 16 consecutive K
            const __bf16* bp = W + (size_t)(n0 + s * 16 + l16) * ldb + k0 + hi * 16;
            v16bf bf = *(const v16bf*)bp;            // 32B aligned
#if defined(__gfx1250__)
            acc[s] = __builtin_amdgcn_wmma_f32_16x16x32_bf16(
                         false, af, false, bf, (short)0, acc[s], false, false);
#else
            (void)bf; (void)af;
#endif
        }
    }

    #pragma unroll
    for (int s = 0; s < 4; ++s) {
        const int n = n0 + s * 16 + l16;
        const float bv = bias ? bias[n] : 0.0f;
        #pragma unroll
        for (int v = 0; v < 8; ++v) {
            const int m = m0 + v + hi * 8;           // C/D layout: VGPR v -> row
            float val = acc[s][v] + bv;
            if (addRow) val += addRow[(size_t)m * ldAdd + n];
            C[(size_t)m * ldc + n] = val;
        }
    }
}

// ---------------------------------------------------------------------------
// LSTM cell elementwise: gates g=[B,4096] in i,f,g,o order. Updates c,h (f32)
// and writes bf16 h into up to two slots of the concat buffers + ys stream.
// ---------------------------------------------------------------------------
__global__ __launch_bounds__(256) void lstm_cell_kernel(
    const float* __restrict__ g,
    float* __restrict__ c, float* __restrict__ h,
    __bf16* __restrict__ hbf1, int ld1, int off1,
    __bf16* __restrict__ hbf2, int ld2, int off2,
    __bf16* __restrict__ ys)
{
    const int idx = blockIdx.x * 256 + threadIdx.x;
    if (idx >= BB * NH_) return;
    const int b = idx >> 10, n = idx & (NH_ - 1);
    const float* gb = g + (size_t)b * H4;
    const float gi = gb[n];
    const float gf = gb[NH_ + n];
    const float gg = gb[2 * NH_ + n];
    const float go = gb[3 * NH_ + n];
    const float si = 1.0f / (1.0f + expf(-gi));
    const float sf = 1.0f / (1.0f + expf(-gf));
    const float so = 1.0f / (1.0f + expf(-go));
    const float cn = sf * c[idx] + si * tanhf(gg);
    const float hn = so * tanhf(cn);
    c[idx] = cn;
    h[idx] = hn;
    const __bf16 hb = (__bf16)hn;
    hbf1[(size_t)b * ld1 + off1 + n] = hb;
    if (hbf2) hbf2[(size_t)b * ld2 + off2 + n] = hb;
    if (ys)   ys[idx] = hb;
}

// ---------------------------------------------------------------------------
// Attention for one decoder step. One block per batch element, 128 threads
// (one per time step). e_t = sum_n tanh(hid_proj[t,b,:]+q[b,:])*att_w, masked
// softmax over t<=tstep, state = (sum softmax) * he1 -> bf16 into xcat0[:,0:NH].
// ---------------------------------------------------------------------------
__global__ __launch_bounds__(128) void attn_kernel(
    const float* __restrict__ hid_proj,  // [T*B, NH]
    const float* __restrict__ q,         // [B, NH]
    const float* __restrict__ att_w,     // [NH]
    const float* __restrict__ he1,       // [B, NH]
    int tstep,
    __bf16* __restrict__ xcat0)          // [B, 2*NH], cols 0..NH-1
{
    __shared__ float sm[TT];
    const int b = blockIdx.x;
    const int t = threadIdx.x;

    float e = -1e30f;
    if (t <= tstep) {
        const float* hp = hid_proj + ((size_t)t * BB + b) * NH_;
        const float* qb = q + (size_t)b * NH_;
        float acc = 0.0f;
        for (int n = 0; n < NH_; ++n)
            acc += tanhf(hp[n] + qb[n]) * att_w[n];
        e = acc;
    }
    sm[t] = e; __syncthreads();
    for (int off = TT / 2; off > 0; off >>= 1) {
        if (t < off) sm[t] = fmaxf(sm[t], sm[t + off]);
        __syncthreads();
    }
    const float mx = sm[0]; __syncthreads();
    const float ex = expf(e - mx);
    sm[t] = ex; __syncthreads();
    for (int off = TT / 2; off > 0; off >>= 1) {
        if (t < off) sm[t] += sm[t + off];
        __syncthreads();
    }
    const float Z = sm[0]; __syncthreads();
    const float en = ex / Z;
    sm[t] = en; __syncthreads();
    for (int off = TT / 2; off > 0; off >>= 1) {
        if (t < off) sm[t] += sm[t + off];
        __syncthreads();
    }
    const float coef = sm[0];   // == 1 analytically (softmax sums to one)
    for (int n = t; n < NH_; n += TT)
        xcat0[(size_t)b * (2 * NH_) + n] = (__bf16)(coef * he1[(size_t)b * NH_ + n]);
}

// ---------------------------------------------------------------------------
// Small helpers
// ---------------------------------------------------------------------------
__global__ void embed_kernel(const int* __restrict__ inp,
                             const float* __restrict__ emb,
                             __bf16* __restrict__ out)
{
    const int i = blockIdx.x * 256 + threadIdx.x;    // < T*B*NI
    if (i >= TT * BB * NI_) return;
    const int tb = i >> 9, k = i & (NI_ - 1);
    out[i] = (__bf16)emb[(size_t)inp[tb] * NI_ + k];
}

__global__ void cvt_kernel(const float* __restrict__ src, int srcLd, int srcOff,
                           __bf16* __restrict__ dst, int dstLd, int dstOff,
                           long long total, int cols)
{
    const long long i = (long long)blockIdx.x * 256 + threadIdx.x;
    if (i >= total) return;
    const int r = (int)(i / cols), c = (int)(i % cols);
    dst[(size_t)r * dstLd + dstOff + c] = (__bf16)src[(size_t)r * srcLd + srcOff + c];
}

__global__ void vecadd_kernel(const float* __restrict__ a,
                              const float* __restrict__ b,
                              float* __restrict__ dst, int n)
{
    const int i = blockIdx.x * 256 + threadIdx.x;
    if (i < n) dst[i] = a[i] + b[i];
}

__global__ void zero_kernel(unsigned int* __restrict__ p, long long n)
{
    const long long i = (long long)blockIdx.x * 256 + threadIdx.x;
    if (i < n) p[i] = 0u;
}

__global__ void pack_hidden_kernel(const float* he0, const float* he1,
                                   const float* hd0, const float* hd1,
                                   const float* ce0, const float* ce1,
                                   const float* cd0, const float* cd1,
                                   float* __restrict__ out)
{
    const int i = blockIdx.x * 256 + threadIdx.x;    // < 8 * B*NH
    if (i >= 8 * BB * NH_) return;
    const int seg = i >> 15, off = i & (BB * NH_ - 1);
    const float* srcs[8] = { he0, he1, hd0, hd1, ce0, ce1, cd0, cd1 };
    out[i] = srcs[seg][off];
}

// ---------------------------------------------------------------------------
// Host orchestration
// ---------------------------------------------------------------------------
static inline dim3 gemm_grid(int M, int N) { return dim3(N / 256, M / 32); }

extern "C" void kernel_launch(void* const* d_in, const int* in_sizes, int n_in,
                              void* d_out, int out_size, void* d_ws, size_t ws_size,
                              hipStream_t stream)
{
    (void)in_sizes; (void)n_in; (void)out_size; (void)ws_size;

    // ---- inputs (setup_inputs order) ----
    const int*   input     = (const int*)  d_in[0];
    const float* embedding = (const float*)d_in[1];
    const float* eWih0 = (const float*)d_in[2];
    const float* eWhh0 = (const float*)d_in[3];
    const float* ebih0 = (const float*)d_in[4];
    const float* ebhh0 = (const float*)d_in[5];
    const float* eWih1 = (const float*)d_in[6];
    const float* eWhh1 = (const float*)d_in[7];
    const float* ebih1 = (const float*)d_in[8];
    const float* ebhh1 = (const float*)d_in[9];
    const float* dWih0 = (const float*)d_in[10];
    const float* dWhh0 = (const float*)d_in[11];
    const float* dbih0 = (const float*)d_in[12];
    const float* dbhh0 = (const float*)d_in[13];
    const float* dWih1 = (const float*)d_in[14];
    const float* dWhh1 = (const float*)d_in[15];
    const float* dbih1 = (const float*)d_in[16];
    const float* dbhh1 = (const float*)d_in[17];
    const float* attWh = (const float*)d_in[18];
    const float* attbh = (const float*)d_in[19];
    const float* attWi = (const float*)d_in[20];
    const float* attbi = (const float*)d_in[21];
    const float* attw  = (const float*)d_in[22];
    const float* outW  = (const float*)d_in[23];
    const float* outb  = (const float*)d_in[24];

    float* logits = (float*)d_out;                         // [T*B, V]
    float* hidden_out = logits + (size_t)TT * BB * VV;     // 8 * [B,NH]

    // ---- workspace carve (deterministic) ----
    char*  base = (char*)d_ws;
    size_t off  = 0;
    auto carve = [&](size_t bytes) -> char* {
        off = (off + 255) & ~(size_t)255;
        char* r = base + off;
        off += bytes;
        return r;
    };
    __bf16* Wih0e   = (__bf16*)carve((size_t)H4 * NI_ * 2);
    __bf16* Whh0e   = (__bf16*)carve((size_t)H4 * NH_ * 2);
    __bf16* Wih1e   = (__bf16*)carve((size_t)H4 * NH_ * 2);
    __bf16* Whh1e   = (__bf16*)carve((size_t)H4 * NH_ * 2);
    __bf16* Wemb0d  = (__bf16*)carve((size_t)H4 * NI_ * 2);
    __bf16* Wcat0   = (__bf16*)carve((size_t)H4 * 2 * NH_ * 2);  // [Wstate|Whh0]
    __bf16* Wcat1   = (__bf16*)carve((size_t)H4 * 2 * NH_ * 2);  // [Wih1 |Whh1]
    __bf16* attWhb  = (__bf16*)carve((size_t)NH_ * NH_ * 2);
    __bf16* attWib  = (__bf16*)carve((size_t)NH_ * NH_ * 2);
    __bf16* outWb   = (__bf16*)carve((size_t)VV * NH_ * 2);
    __bf16* emb_bf  = (__bf16*)carve((size_t)TT * BB * NI_ * 2);
    __bf16* ys0_bf  = (__bf16*)carve((size_t)TT * BB * NH_ * 2);
    __bf16* enc_bf  = (__bf16*)carve((size_t)TT * BB * NH_ * 2);
    __bf16* preds_bf= (__bf16*)carve((size_t)TT * BB * NH_ * 2);
    float*  bufXp   = (float*)carve((size_t)TT * BB * H4 * 4);   // Xproj0 then Xproj1
    float*  bufEmbP = (float*)carve((size_t)TT * BB * H4 * 4);   // decoder emb proj
    float*  hidproj = (float*)carve((size_t)TT * BB * NH_ * 4);
    float*  gbuf    = (float*)carve((size_t)BB * H4 * 4);
    float*  qbuf    = (float*)carve((size_t)BB * NH_ * 4);
    float*  bsum0e  = (float*)carve((size_t)H4 * 4);
    float*  bsum1e  = (float*)carve((size_t)H4 * 4);
    float*  bsum0d  = (float*)carve((size_t)H4 * 4);
    float*  bsum1d  = (float*)carve((size_t)H4 * 4);
    // zero-initialized region: 8 f32 state buffers + bf16 h/concat buffers
    const size_t ZDW = 8 * (size_t)BB * NH_                 // f32 words
                     + (2 * (size_t)BB * NH_ + 2 * (size_t)BB * 2 * NH_) / 2;
    float* zreg = (float*)carve(ZDW * 4);
    float* h_e0 = zreg;             float* c_e0 = zreg + 1 * BB * NH_;
    float* h_e1 = zreg + 2*BB*NH_;  float* c_e1 = zreg + 3 * BB * NH_;
    float* h_d0 = zreg + 4*BB*NH_;  float* c_d0 = zreg + 5 * BB * NH_;
    float* h_d1 = zreg + 6*BB*NH_;  float* c_d1 = zreg + 7 * BB * NH_;
    __bf16* hbf_e0 = (__bf16*)(zreg + 8 * BB * NH_);
    __bf16* hbf_e1 = hbf_e0 + BB * NH_;
    __bf16* xcat0  = hbf_e1 + BB * NH_;      // [B, 2NH] = [state | h_d0]
    __bf16* xcat1  = xcat0 + BB * 2 * NH_;   // [B, 2NH] = [h_d0  | h_d1]

    const dim3 blk(256);
    auto cblocks = [](long long total) { return dim3((unsigned)((total + 255) / 256)); };

    // ---- weight conversion to bf16 (+ concat layouts) ----
    cvt_kernel<<<cblocks((long long)H4*NI_), blk, 0, stream>>>(eWih0, NI_, 0, Wih0e, NI_, 0, (long long)H4*NI_, NI_);
    cvt_kernel<<<cblocks((long long)H4*NH_), blk, 0, stream>>>(eWhh0, NH_, 0, Whh0e, NH_, 0, (long long)H4*NH_, NH_);
    cvt_kernel<<<cblocks((long long)H4*NH_), blk, 0, stream>>>(eWih1, NH_, 0, Wih1e, NH_, 0, (long long)H4*NH_, NH_);
    cvt_kernel<<<cblocks((long long)H4*NH_), blk, 0, stream>>>(eWhh1, NH_, 0, Whh1e, NH_, 0, (long long)H4*NH_, NH_);
    // dec_Wih0 [4096, 1536]: cols 0:1024 = state part, cols 1024:1536 = emb part
    cvt_kernel<<<cblocks((long long)H4*NH_), blk, 0, stream>>>(dWih0, NH_+NI_, 0,    Wcat0, 2*NH_, 0,    (long long)H4*NH_, NH_);
    cvt_kernel<<<cblocks((long long)H4*NH_), blk, 0, stream>>>(dWhh0, NH_,     0,    Wcat0, 2*NH_, NH_,  (long long)H4*NH_, NH_);
    cvt_kernel<<<cblocks((long long)H4*NI_), blk, 0, stream>>>(dWih0, NH_+NI_, NH_,  Wemb0d, NI_,  0,    (long long)H4*NI_, NI_);
    cvt_kernel<<<cblocks((long long)H4*NH_), blk, 0, stream>>>(dWih1, NH_,     0,    Wcat1, 2*NH_, 0,    (long long)H4*NH_, NH_);
    cvt_kernel<<<cblocks((long long)H4*NH_), blk, 0, stream>>>(dWhh1, NH_,     0,    Wcat1, 2*NH_, NH_,  (long long)H4*NH_, NH_);
    cvt_kernel<<<cblocks((long long)NH_*NH_), blk, 0, stream>>>(attWh, NH_, 0, attWhb, NH_, 0, (long long)NH_*NH_, NH_);
    cvt_kernel<<<cblocks((long long)NH_*NH_), blk, 0, stream>>>(attWi, NH_, 0, attWib, NH_, 0, (long long)NH_*NH_, NH_);
    cvt_kernel<<<cblocks((long long)VV*NH_),  blk, 0, stream>>>(outW,  NH_, 0, outWb,  NH_, 0, (long long)VV*NH_,  NH_);

    // ---- bias sums (bih + bhh folded into input projections) ----
    vecadd_kernel<<<cblocks(H4), blk, 0, stream>>>(ebih0, ebhh0, bsum0e, H4);
    vecadd_kernel<<<cblocks(H4), blk, 0, stream>>>(ebih1, ebhh1, bsum1e, H4);
    vecadd_kernel<<<cblocks(H4), blk, 0, stream>>>(dbih0, dbhh0, bsum0d, H4);
    vecadd_kernel<<<cblocks(H4), blk, 0, stream>>>(dbih1, dbhh1, bsum1d, H4);

    // ---- zero h/c/concat state ----
    zero_kernel<<<cblocks((long long)ZDW), blk, 0, stream>>>((unsigned int*)zreg, (long long)ZDW);

    // ---- embedding gather -> bf16 ----
    embed_kernel<<<cblocks((long long)TT*BB*NI_), blk, 0, stream>>>(input, embedding, emb_bf);

    const int M_seq = TT * BB;   // 4096

    // ---- big input-projection GEMMs (parallel over all timesteps) ----
    // Xproj0 = emb @ Wih0e^T + (bih0+bhh0)
    gemm_bf16_wmma<<<gemm_grid(M_seq, H4), blk, 0, stream>>>(
        emb_bf, NI_, Wih0e, NI_, bsum0e, nullptr, 0, bufXp, H4, M_seq, H4, NI_);
    // EmbProj = emb @ Wemb0d^T + (dbih0+dbhh0)
    gemm_bf16_wmma<<<gemm_grid(M_seq, H4), blk, 0, stream>>>(
        emb_bf, NI_, Wemb0d, NI_, bsum0d, nullptr, 0, bufEmbP, H4, M_seq, H4, NI_);

    // ---- encoder layer 0 (sequential over T) ----
    for (int t = 0; t < TT; ++t) {
        gemm_bf16_wmma<<<gemm_grid(BB, H4), blk, 0, stream>>>(
            hbf_e0, NH_, Whh0e, NH_, nullptr,
            bufXp + (size_t)t * BB * H4, H4, gbuf, H4, BB, H4, NH_);
        lstm_cell_kernel<<<dim3(BB * NH_ / 256), blk, 0, stream>>>(
            gbuf, c_e0, h_e0, hbf_e0, NH_, 0, nullptr, 0, 0,
            ys0_bf + (size_t)t * BB * NH_);
    }

    // ---- encoder layer 1 input projection (reuse bufXp) ----
    gemm_bf16_wmma<<<gemm_grid(M_seq, H4), blk, 0, stream>>>(
        ys0_bf, NH_, Wih1e, NH_, bsum1e, nullptr, 0, bufXp, H4, M_seq, H4, NH_);

    // ---- encoder layer 1 (sequential) ----
    for (int t = 0; t < TT; ++t) {
        gemm_bf16_wmma<<<gemm_grid(BB, H4), blk, 0, stream>>>(
            hbf_e1, NH_, Whh1e, NH_, nullptr,
            bufXp + (size_t)t * BB * H4, H4, gbuf, H4, BB, H4, NH_);
        lstm_cell_kernel<<<dim3(BB * NH_ / 256), blk, 0, stream>>>(
            gbuf, c_e1, h_e1, hbf_e1, NH_, 0, nullptr, 0, 0,
            enc_bf + (size_t)t * BB * NH_);
    }

    // ---- attention keys: hid_proj = enc_out @ attWh^T + attbh ----
    gemm_bf16_wmma<<<gemm_grid(M_seq, NH_), blk, 0, stream>>>(
        enc_bf, NH_, attWhb, NH_, attbh, nullptr, 0, hidproj, NH_, M_seq, NH_, NH_);

    // ---- decoder (sequential) ----
    for (int t = 0; t < TT; ++t) {
        // q = h1_prev @ attWi^T + attbi   (h1_prev lives in xcat1[:, NH:2NH])
        gemm_bf16_wmma<<<gemm_grid(BB, NH_), blk, 0, stream>>>(
            xcat1 + NH_, 2 * NH_, attWib, NH_, attbi, nullptr, 0,
            qbuf, NH_, BB, NH_, NH_);
        // masked softmax attention -> state (bf16 into xcat0[:, 0:NH])
        attn_kernel<<<dim3(BB), dim3(TT), 0, stream>>>(
            hidproj, qbuf, attw, h_e1, t, xcat0);
        // g0 = [state|h0] @ [Wstate|Whh0]^T + EmbProj[t]
        gemm_bf16_wmma<<<gemm_grid(BB, H4), blk, 0, stream>>>(
            xcat0, 2 * NH_, Wcat0, 2 * NH_, nullptr,
            bufEmbP + (size_t)t * BB * H4, H4, gbuf, H4, BB, H4, 2 * NH_);
        // cell0: writes h0 -> xcat0[:,NH:] (for next step) and xcat1[:,0:NH]
        lstm_cell_kernel<<<dim3(BB * NH_ / 256), blk, 0, stream>>>(
            gbuf, c_d0, h_d0, xcat0, 2 * NH_, NH_, xcat1, 2 * NH_, 0, nullptr);
        // g1 = [h0_new|h1_prev] @ [Wih1|Whh1]^T + (dbih1+dbhh1)
        gemm_bf16_wmma<<<gemm_grid(BB, H4), blk, 0, stream>>>(
            xcat1, 2 * NH_, Wcat1, 2 * NH_, bsum1d, nullptr, 0,
            gbuf, H4, BB, H4, 2 * NH_);
        // cell1: writes h1 -> xcat1[:,NH:] and preds stream
        lstm_cell_kernel<<<dim3(BB * NH_ / 256), blk, 0, stream>>>(
            gbuf, c_d1, h_d1, xcat1, 2 * NH_, NH_, nullptr, 0, 0,
            preds_bf + (size_t)t * BB * NH_);
    }

    // ---- logits = preds @ out_W^T + out_b  (dominant GEMM, 268 GFLOP) ----
    gemm_bf16_wmma<<<gemm_grid(M_seq, VV), blk, 0, stream>>>(
        preds_bf, NH_, outWb, NH_, outb, nullptr, 0, logits, VV, M_seq, VV, NH_);

    // ---- pack hidden state tuple: [he0,he1,hd0,hd1, ce0,ce1,cd0,cd1] ----
    pack_hidden_kernel<<<cblocks(8LL * BB * NH_), blk, 0, stream>>>(
        h_e0, h_e1, h_d0, h_d1, c_e0, c_e1, c_d0, c_d1, hidden_out);
}
